// HardNegativeCrossAttention_32074815767112
// MI455X (gfx1250) — matrix operand order repaired and checked by reading, not compiled, not run on hardware
//
#include <hip/hip_runtime.h>
#include <hip/hip_bf16.h>
#include <stdint.h>

// ---------------- problem constants ----------------
constexpr int cB  = 8;
constexpr int cC  = 256;
constexpr int cH  = 56;
constexpr int cW  = 56;
constexpr int cHW = cH * cW;        // 3136
constexpr int cM  = cB * cHW;       // 25088 tokens
constexpr int cR  = 16;

typedef __bf16 bf16_t;
typedef bf16_t v16bf __attribute__((ext_vector_type(16)));
typedef float  v8f   __attribute__((ext_vector_type(8)));

struct alignas(16) U4 { unsigned int x, y, z, w; };
struct alignas(16) F4 { float x, y, z, w; };

union Frag { U4 q[2]; v16bf v; unsigned short s[16]; };

__device__ __forceinline__ unsigned short f2bf(float f) {
  union { float f; unsigned u; } v; v.f = f;
  unsigned r = v.u + 0x7FFFu + ((v.u >> 16) & 1u);   // round-to-nearest-even
  return (unsigned short)(r >> 16);
}

// =====================================================================
// Generic bf16 WMMA GEMM:   O[m][n] (+)= alpha * sum_k A[m][k]*B[n][k]
//   MODE_B 0: B = f32 weights, row-major [Nout][ldb], column offset koffB
//   MODE_B 1: B = bf16 [Nout][ldb]              (used for S = Q*K^T)
//   MODE_B 2: B = bf16 [K][ldb=Nout] (transposed staging, used for P*V)
// Tile: 128x64, 8 waves, each wave 32x32 (2x2 WMMA 16x16x32 tiles).
// =====================================================================
template<int MODE_B, bool RELU, bool ACCUM, bool BIAS, bool MULAUX,
         bool OUTF, bool OUTB>
__launch_bounds__(256)
__global__ void gemm_wmma(const unsigned short* __restrict__ A,
                          const void* __restrict__ Bsrc,
                          const float* __restrict__ bias,
                          const float* __restrict__ aux,
                          float* __restrict__ Of,
                          unsigned short* __restrict__ Ob,
                          int M, int Nn, int K, int lda, int ldb, int koffB,
                          float alpha)
{
  __shared__ unsigned short sA[2][128][48];   // 48 = 32 + pad, keeps 16B align
  __shared__ unsigned short sB[2][64][48];

  const int tid  = threadIdx.x;
  const int m0   = blockIdx.y * 128;
  const int n0   = blockIdx.x * 64;
  const int lane = tid & 31;
  const int hf   = lane >> 4;      // lane half (0/1)
  const int r    = lane & 15;
  const int wv   = tid >> 5;
  const int wr   = wv & 3;         // wave row group (4)
  const int wc   = wv >> 2;        // wave col group (2)

  v8f acc[2][2];
  #pragma unroll
  for (int i = 0; i < 2; ++i)
    #pragma unroll
    for (int j = 0; j < 2; ++j)
      #pragma unroll
      for (int g = 0; g < 8; ++g) acc[i][j][g] = 0.f;

  auto stage = [&](int buf, int kt) {
    // ---- A tile: 128 x 32 bf16 (two uint4 per thread) ----
    #pragma unroll
    for (int it = 0; it < 2; ++it) {
      int v   = tid + it * 256;
      int row = v >> 2;
      int kq  = v & 3;
      U4 val = {0u, 0u, 0u, 0u};
      int gm = m0 + row;
      if (gm < M)
        val = *(const U4*)(A + (size_t)gm * lda + (size_t)kt * 32 + kq * 8);
      *(U4*)&sA[buf][row][kq * 8] = val;
    }
    // ---- B tile staged transposed-to-[n][k] ----
    if constexpr (MODE_B == 0) {
      const float* Wt = (const float*)Bsrc;
      int n = tid >> 2, kq = tid & 3;
      const float* p = Wt + (size_t)(n0 + n) * ldb + koffB + (size_t)kt * 32 + kq * 8;
      F4 f0 = *(const F4*)p;
      F4 f1 = *(const F4*)(p + 4);
      unsigned short* d = &sB[buf][n][kq * 8];
      d[0] = f2bf(f0.x); d[1] = f2bf(f0.y); d[2] = f2bf(f0.z); d[3] = f2bf(f0.w);
      d[4] = f2bf(f1.x); d[5] = f2bf(f1.y); d[6] = f2bf(f1.z); d[7] = f2bf(f1.w);
    } else if constexpr (MODE_B == 1) {
      const unsigned short* Bt = (const unsigned short*)Bsrc;
      int n = tid >> 2, kq = tid & 3;
      *(U4*)&sB[buf][n][kq * 8] =
          *(const U4*)(Bt + (size_t)(n0 + n) * ldb + (size_t)kt * 32 + kq * 8);
    } else {
      const unsigned short* Bt = (const unsigned short*)Bsrc;
      int kk = tid >> 3, nq = tid & 7;          // 32 k-rows x 8 col-groups
      U4 val = *(const U4*)(Bt + (size_t)((size_t)kt * 32 + kk) * ldb + n0 + nq * 8);
      union { U4 q; unsigned short s[8]; } u; u.q = val;
      #pragma unroll
      for (int e = 0; e < 8; ++e) sB[buf][nq * 8 + e][kk] = u.s[e];
    }
  };

  auto compute = [&](int buf) {
    Frag fa[2], fb[2];
    #pragma unroll
    for (int i = 0; i < 2; ++i) {
      // 16-bit A 16x32 layout: vgpr0..3 <- K = hf*8 + 0..7 ; vgpr4..7 <- K = 16+hf*8+0..7
      const unsigned short* pa = &sA[buf][wr * 32 + i * 16 + r][hf * 8];
      fa[i].q[0] = *(const U4*)pa;
      fa[i].q[1] = *(const U4*)(pa + 16);
    }
    #pragma unroll
    for (int j = 0; j < 2; ++j) {
      const unsigned short* pb = &sB[buf][wc * 32 + j * 16 + r][hf * 8];
      fb[j].q[0] = *(const U4*)pb;
      fb[j].q[1] = *(const U4*)(pb + 16);
    }
    #pragma unroll
    for (int i = 0; i < 2; ++i)
      #pragma unroll
      for (int j = 0; j < 2; ++j)
        acc[i][j] = __builtin_amdgcn_wmma_f32_16x16x32_bf16(
            false, fa[i].v, false, fb[j].v, (short)0, acc[i][j], false, false);
  };

  const int nk = K >> 5;
  stage(0, 0);
  __syncthreads();
  int buf = 0;
  for (int kt = 0; kt < nk; ++kt) {
    if (kt + 2 < nk) {                       // L2 prefetch of tile after next
      int gm = m0 + (tid & 127);
      if (gm < M)
        __builtin_prefetch(A + (size_t)gm * lda + (size_t)(kt + 2) * 32, 0, 1);
    }
    if (kt + 1 < nk) stage(buf ^ 1, kt + 1); // writes other buffer
    compute(buf);
    __syncthreads();
    buf ^= 1;
  }

  // ---- epilogue ----
  #pragma unroll
  for (int i = 0; i < 2; ++i) {
    #pragma unroll
    for (int j = 0; j < 2; ++j) {
      int col = n0 + wc * 32 + j * 16 + r;
      #pragma unroll
      for (int g = 0; g < 8; ++g) {
        int row = m0 + wr * 32 + i * 16 + hf * 8 + g;
        if (row < M) {
          float v = acc[i][j][g] * alpha;
          size_t oidx = (size_t)row * Nn + col;
          if constexpr (BIAS)   v += bias[col];
          if constexpr (ACCUM)  v += Of[oidx];
          if constexpr (RELU)   v = fmaxf(v, 0.f);
          if constexpr (MULAUX) v *= aux[oidx];
          if constexpr (OUTF)   Of[oidx] = v;
          if constexpr (OUTB)   Ob[oidx] = f2bf(v);
        }
      }
    }
  }
}

// =====================================================================
// Implicit-GEMM 3x3 conv / conv-transpose on token-major [B,H,W,C] bf16.
// K = 9*256 = 2304, k = tap*256 + c  (so each 32-wide K block is one tap).
//   CONVT=false: W = [O][C][3][3]         (corr, pad 1)
//   CONVT=true : W = [C][O][3][3], taps flipped (ConvTranspose2d, pad 1)
// =====================================================================
template<bool CONVT, bool RELU, bool MULAUX>
__launch_bounds__(256)
__global__ void conv3x3_wmma(const unsigned short* __restrict__ Atok,
                             const float* __restrict__ Wt,
                             const float* __restrict__ bias,
                             const float* __restrict__ aux,
                             unsigned short* __restrict__ Ob)
{
  __shared__ unsigned short sA[2][128][48];
  __shared__ unsigned short sB[2][64][48];

  const int tid  = threadIdx.x;
  const int m0   = blockIdx.y * 128;
  const int n0   = blockIdx.x * 64;
  const int lane = tid & 31;
  const int hf   = lane >> 4;
  const int r    = lane & 15;
  const int wv   = tid >> 5;
  const int wr   = wv & 3;
  const int wc   = wv >> 2;

  v8f acc[2][2];
  #pragma unroll
  for (int i = 0; i < 2; ++i)
    #pragma unroll
    for (int j = 0; j < 2; ++j)
      #pragma unroll
      for (int g = 0; g < 8; ++g) acc[i][j][g] = 0.f;

  auto stage = [&](int buf, int kt) {
    #pragma unroll
    for (int it = 0; it < 2; ++it) {
      int v   = tid + it * 256;
      int row = v >> 2;
      int kq  = v & 3;
      int k   = kt * 32 + kq * 8;
      int tap = k >> 8;            // 0..8
      int c0  = k & 255;
      int gm  = m0 + row;
      U4 val = {0u, 0u, 0u, 0u};
      if (gm < cM) {
        int b  = gm / cHW;
        int hw = gm - b * cHW;
        int y  = hw / cW;
        int x  = hw - y * cW;
        int ys = y + tap / 3 - 1;
        int xs = x + tap % 3 - 1;
        if (ys >= 0 && ys < cH && xs >= 0 && xs < cW)
          val = *(const U4*)(Atok + (size_t)((b * cH + ys) * cW + xs) * cC + c0);
      }
      *(U4*)&sA[buf][row][kq * 8] = val;
    }
    int n = tid >> 2, kq = tid & 3;
    #pragma unroll
    for (int e = 0; e < 8; ++e) {
      int k   = kt * 32 + kq * 8 + e;
      int tap = k >> 8;
      int c   = k & 255;
      size_t widx = CONVT ? ((size_t)c * cC + (n0 + n)) * 9 + (8 - tap)
                          : ((size_t)(n0 + n) * cC + c) * 9 + tap;
      sB[buf][n][kq * 8 + e] = f2bf(Wt[widx]);
    }
  };

  auto compute = [&](int buf) {
    Frag fa[2], fb[2];
    #pragma unroll
    for (int i = 0; i < 2; ++i) {
      const unsigned short* pa = &sA[buf][wr * 32 + i * 16 + r][hf * 8];
      fa[i].q[0] = *(const U4*)pa;
      fa[i].q[1] = *(const U4*)(pa + 16);
    }
    #pragma unroll
    for (int j = 0; j < 2; ++j) {
      const unsigned short* pb = &sB[buf][wc * 32 + j * 16 + r][hf * 8];
      fb[j].q[0] = *(const U4*)pb;
      fb[j].q[1] = *(const U4*)(pb + 16);
    }
    #pragma unroll
    for (int i = 0; i < 2; ++i)
      #pragma unroll
      for (int j = 0; j < 2; ++j)
        acc[i][j] = __builtin_amdgcn_wmma_f32_16x16x32_bf16(
            false, fa[i].v, false, fb[j].v, (short)0, acc[i][j], false, false);
  };

  const int nk = (cC * 9) >> 5;   // 72
  stage(0, 0);
  __syncthreads();
  int buf = 0;
  for (int kt = 0; kt < nk; ++kt) {
    if (kt + 1 < nk) stage(buf ^ 1, kt + 1);
    compute(buf);
    __syncthreads();
    buf ^= 1;
  }

  #pragma unroll
  for (int i = 0; i < 2; ++i) {
    #pragma unroll
    for (int j = 0; j < 2; ++j) {
      int col = n0 + wc * 32 + j * 16 + r;
      #pragma unroll
      for (int g = 0; g < 8; ++g) {
        int row = m0 + wr * 32 + i * 16 + hf * 8 + g;
        if (row < cM) {
          float v = acc[i][j][g] + bias[col];
          size_t oidx = (size_t)row * cC + col;
          if constexpr (RELU)   v = fmaxf(v, 0.f);
          if constexpr (MULAUX) v *= aux[oidx];
          Ob[oidx] = f2bf(v);
        }
      }
    }
  }
}

// ---------------- small helper kernels ----------------
__global__ void nchw_to_tok(const float* __restrict__ src,
                            unsigned short* __restrict__ dst)
{
  int i = blockIdx.x * 256 + threadIdx.x;           // NCHW element index
  if (i >= cM * cC) return;
  int hw = i % cHW;
  int t  = i / cHW;
  int c  = t % cC;
  int b  = t / cC;
  dst[(size_t)(b * cHW + hw) * cC + c] = f2bf(src[i]);
}

__global__ void f32_to_bf(const float* __restrict__ s,
                          unsigned short* __restrict__ d, int n)
{
  int i = blockIdx.x * 256 + threadIdx.x;
  if (i < n) d[i] = f2bf(s[i]);
}

__global__ void zero_f32(float* __restrict__ p, int n)
{
  int i = blockIdx.x * 256 + threadIdx.x;
  if (i < n) p[i] = 0.f;
}

// attr branch stage 1: a1[m][r] = relu(b1[r] + sum_c w1[r][c]*x[b,c,hw])
__global__ void attr1_kernel(const float* __restrict__ x,
                             const float* __restrict__ w1,
                             const float* __restrict__ b1,
                             float* __restrict__ a1)
{
  __shared__ float xs[16][257];
  int tid  = threadIdx.x;
  int m0   = blockIdx.x * 16;
  int ti   = tid >> 4;
  int part = tid & 15;
  int m    = m0 + ti;
  int b    = m / cHW;
  int hw   = m - b * cHW;
  const float* xb = x + (size_t)b * cC * cHW + hw;
  #pragma unroll
  for (int e = 0; e < 16; ++e) {
    int c = part * 16 + e;
    xs[ti][c] = xb[(size_t)c * cHW];
  }
  __syncthreads();
  int rr = part;
  float accv = b1[rr];
  const float* wr_ = w1 + rr * cC;
  for (int c = 0; c < cC; ++c) accv += wr_[c] * xs[ti][c];
  a1[(size_t)m * cR + rr] = fmaxf(accv, 0.f);
}

// attr branch stage 2: a2[m][c] = relu(b2[c] + sum_r w2[c][r]*a1[m][r])
__global__ void attr2_kernel(const float* __restrict__ a1,
                             const float* __restrict__ w2,
                             const float* __restrict__ b2,
                             unsigned short* __restrict__ a2bf)
{
  int i = blockIdx.x * 256 + threadIdx.x;
  if (i >= cM * cC) return;
  int m = i >> 8, c = i & 255;
  float acc = b2[c];
  const float* w  = w2 + c * cR;
  const float* av = a1 + (size_t)m * cR;
  #pragma unroll
  for (int rr = 0; rr < cR; ++rr) acc += w[rr] * av[rr];
  a2bf[i] = f2bf(fmaxf(acc, 0.f));
}

// row softmax over pre-scaled scores, writes bf16 probabilities
__global__ void softmax_rows(const float* __restrict__ S,
                             unsigned short* __restrict__ P, int n)
{
  __shared__ float red[256];
  int row = blockIdx.x, tid = threadIdx.x;
  const float* s = S + (size_t)row * n;
  float m = -3.4e38f;
  for (int c = tid; c < n; c += 256) m = fmaxf(m, s[c]);
  red[tid] = m; __syncthreads();
  for (int o = 128; o > 0; o >>= 1) {
    if (tid < o) red[tid] = fmaxf(red[tid], red[tid + o]);
    __syncthreads();
  }
  m = red[0]; __syncthreads();
  float sum = 0.f;
  for (int c = tid; c < n; c += 256) sum += __expf(s[c] - m);
  red[tid] = sum; __syncthreads();
  for (int o = 128; o > 0; o >>= 1) {
    if (tid < o) red[tid] += red[tid + o];
    __syncthreads();
  }
  float inv = 1.0f / red[0];
  unsigned short* p = P + (size_t)row * n;
  for (int c = tid; c < n; c += 256) p[c] = f2bf(__expf(s[c] - m) * inv);
}

// BatchNorm stats over residual sum (one block per channel)
__global__ void bn_stats(const float* __restrict__ o_tok,
                         const float* __restrict__ x,
                         float* __restrict__ stats)
{
  __shared__ float rs[256], rq[256];
  int c = blockIdx.x, tid = threadIdx.x;
  float s = 0.f, q = 0.f;
  for (int m = tid; m < cM; m += 256) {
    int b = m / cHW, hw = m - b * cHW;
    float v = o_tok[(size_t)m * cC + c] + x[((size_t)b * cC + c) * cHW + hw];
    s += v; q += v * v;
  }
  rs[tid] = s; rq[tid] = q; __syncthreads();
  for (int o = 128; o > 0; o >>= 1) {
    if (tid < o) { rs[tid] += rs[tid + o]; rq[tid] += rq[tid + o]; }
    __syncthreads();
  }
  if (tid == 0) {
    float mean = rs[0] / (float)cM;
    float var  = rq[0] / (float)cM - mean * mean;
    stats[c]      = mean;
    stats[cC + c] = rsqrtf(var + 1e-5f);
  }
}

__global__ void bn_apply(const float* __restrict__ o_tok,
                         const float* __restrict__ x,
                         const float* __restrict__ stats,
                         const float* __restrict__ g,
                         const float* __restrict__ beta,
                         float* __restrict__ out)
{
  int i = blockIdx.x * 256 + threadIdx.x;           // NCHW index
  if (i >= cB * cC * cHW) return;
  int hw = i % cHW;
  int t  = i / cHW;
  int c  = t % cC;
  int b  = t / cC;
  float v = o_tok[(size_t)(b * cHW + hw) * cC + c] + x[i];
  out[i] = (v - stats[c]) * stats[cC + c] * g[c] + beta[c];
}

// =====================================================================
extern "C" void kernel_launch(void* const* d_in, const int* in_sizes, int n_in,
                              void* d_out, int out_size, void* d_ws, size_t ws_size,
                              hipStream_t stream)
{
  (void)in_sizes; (void)n_in; (void)out_size; (void)ws_size;

  const float* x        = (const float*)d_in[0];
  const float* hard_neg = (const float*)d_in[1];
  const float* attr_c1_w= (const float*)d_in[2];
  const float* attr_c1_b= (const float*)d_in[3];
  const float* attr_c2_w= (const float*)d_in[4];
  const float* attr_c2_b= (const float*)d_in[5];
  const float* adj_w    = (const float*)d_in[6];
  const float* adj_b    = (const float*)d_in[7];
  const float* tex_c1_w = (const float*)d_in[8];
  const float* tex_c1_b = (const float*)d_in[9];
  const float* tex_c2_w = (const float*)d_in[10];
  const float* tex_c2_b = (const float*)d_in[11];
  const float* qkv_w[6] = { (const float*)d_in[12], (const float*)d_in[14],
                            (const float*)d_in[16], (const float*)d_in[18],
                            (const float*)d_in[20], (const float*)d_in[22] };
  const float* qkv_b[6] = { (const float*)d_in[13], (const float*)d_in[15],
                            (const float*)d_in[17], (const float*)d_in[19],
                            (const float*)d_in[21], (const float*)d_in[23] };
  const float* out_w    = (const float*)d_in[24];
  const float* out_b    = (const float*)d_in[25];
  const float* bn_g     = (const float*)d_in[26];
  const float* bn_b     = (const float*)d_in[27];
  float* out = (float*)d_out;

  // ---- workspace carve (~305 MB) ----
  char* wsp = (char*)d_ws;
  auto take = [&](size_t bytes) {
    char* p = wsp;
    wsp += (bytes + 255) & ~(size_t)255;
    return p;
  };
  const size_t tokBF = (size_t)cM * cC * sizeof(unsigned short);
  const size_t tokF  = (size_t)cM * cC * sizeof(float);

  unsigned short* x_tok   = (unsigned short*)take(tokBF);
  unsigned short* hn_tok  = (unsigned short*)take(tokBF);
  float*          a1      = (float*)take((size_t)cM * cR * sizeof(float));
  unsigned short* a2_bf   = (unsigned short*)take(tokBF);
  float*          a_f32   = (float*)take(tokF);
  unsigned short* a_bf    = (unsigned short*)take(tokBF);
  unsigned short* t1_bf   = (unsigned short*)take(tokBF);
  unsigned short* t_bf    = (unsigned short*)take(tokBF);
  unsigned short* QKV[6];                       // Qa Ka Va Qt Kt Vt
  for (int i = 0; i < 6; ++i) QKV[i] = (unsigned short*)take(tokBF);
  float*          Sbuf    = (float*)take((size_t)cHW * cHW * sizeof(float));
  unsigned short* Pbuf    = (unsigned short*)take((size_t)cHW * cHW * sizeof(unsigned short));
  float*          o_attn  = (float*)take(tokF);
  unsigned short* o_at_bf = (unsigned short*)take(tokBF);
  float*          o_tok   = (float*)take(tokF);
  float*          stats   = (float*)take(2 * cC * sizeof(float));

  const int elemBlocks = (cM * cC + 255) / 256;   // 25088
  const dim3 gemmGrid(cC / 64, cM / 128);         // (4, 196)

  // 1) convert inputs to token-major bf16
  nchw_to_tok<<<elemBlocks, 256, 0, stream>>>(x, x_tok);
  nchw_to_tok<<<elemBlocks, 256, 0, stream>>>(hard_neg, hn_tok);

  // 2) attr squeeze/excite (tiny, VALU)
  attr1_kernel<<<cM / 16, 256, 0, stream>>>(x, attr_c1_w, attr_c1_b, a1);
  attr2_kernel<<<elemBlocks, 256, 0, stream>>>(a1, attr_c2_w, attr_c2_b, a2_bf);

  // 3) adjust 1x1 over concat([a2, x]) -> a   (two K passes into [C,2C] weight)
  gemm_wmma<0,false,false,true ,false,true ,false><<<gemmGrid, 256, 0, stream>>>(
      a2_bf, adj_w, adj_b, nullptr, a_f32, nullptr, cM, cC, cC, cC, 2*cC, 0, 1.f);
  gemm_wmma<0,false,true ,false,false,true ,true ><<<gemmGrid, 256, 0, stream>>>(
      x_tok, adj_w, nullptr, nullptr, a_f32, a_bf, cM, cC, cC, cC, 2*cC, cC, 1.f);

  // 4) texture branch: relu(conv3x3) then convT3x3, modulated by a
  conv3x3_wmma<false,true ,false><<<gemmGrid, 256, 0, stream>>>(
      hn_tok, tex_c1_w, tex_c1_b, nullptr, t1_bf);
  conv3x3_wmma<true ,false,true ><<<gemmGrid, 256, 0, stream>>>(
      t1_bf, tex_c2_w, tex_c2_b, a_f32, t_bf);

  // 5) six C x C projections (Qa Ka Va from a, Qt Kt Vt from t)
  for (int i = 0; i < 6; ++i) {
    const unsigned short* src = (i < 3) ? a_bf : t_bf;
    gemm_wmma<0,false,false,true,false,false,true><<<gemmGrid, 256, 0, stream>>>(
        src, qkv_w[i], qkv_b[i], nullptr, nullptr, QKV[i], cM, cC, cC, cC, cC, 0, 1.f);
  }

  // 6) attention, per (batch, branch); S stays resident in 192MB L2
  zero_f32<<<elemBlocks, 256, 0, stream>>>(o_attn, cM * cC);
  const dim3 sGrid(cHW / 64, (cHW + 127) / 128);  // (49, 25)
  const dim3 pvGrid(cC / 64, (cHW + 127) / 128);  // (4, 25)
  const float scale = 1.0f / 16.0f;               // 1/sqrt(256)
  for (int b = 0; b < cB; ++b) {
    for (int br = 0; br < 2; ++br) {
      size_t off = (size_t)b * cHW * cC;
      const unsigned short* Q = QKV[br * 3 + 0] + off;
      const unsigned short* K = QKV[br * 3 + 1] + off;
      const unsigned short* V = QKV[br * 3 + 2] + off;
      gemm_wmma<1,false,false,false,false,true,false><<<sGrid, 256, 0, stream>>>(
          Q, K, nullptr, nullptr, Sbuf, nullptr, cHW, cHW, cC, cC, cC, 0, scale);
      softmax_rows<<<cHW, 256, 0, stream>>>(Sbuf, Pbuf, cHW);
      gemm_wmma<2,false,true,false,false,true,false><<<pvGrid, 256, 0, stream>>>(
          Pbuf, V, nullptr, nullptr, o_attn + off, nullptr,
          cHW, cC, cHW, cHW, cC, 0, 0.5f);      // alpha-blend accumulate
    }
  }

  // 7) output 1x1 conv, then fused residual + BatchNorm (+ NCHW transpose)
  f32_to_bf<<<elemBlocks, 256, 0, stream>>>(o_attn, o_at_bf, cM * cC);
  gemm_wmma<0,false,false,true,false,true,false><<<gemmGrid, 256, 0, stream>>>(
      o_at_bf, out_w, out_b, nullptr, o_tok, nullptr, cM, cC, cC, cC, cC, 0, 1.f);
  bn_stats<<<cC, 256, 0, stream>>>(o_tok, x, stats);
  bn_apply<<<elemBlocks, 256, 0, stream>>>(o_tok, x, stats, bn_g, bn_b, out);
}
